// DConv2dBlock_59468117180728
// MI455X (gfx1250) — matrix operand order repaired and checked by reading, not compiled, not run on hardware
//
#include <hip/hip_runtime.h>
#include <hip/hip_bf16.h>

// Problem constants
#define Bn 16
#define Cc 32
#define Hh 128
#define Ww 128
#define Oo 64
#define Kk 9
#define HW (Hh*Ww)
// LDS row pitch in ushorts: 288 data + 8 pad = 296 (592B, 16B-aligned rows)
#define PITCH 296

typedef __attribute__((ext_vector_type(16))) __bf16       v16bf;
typedef __attribute__((ext_vector_type(8)))  float        v8f;
typedef __attribute__((ext_vector_type(4)))  unsigned int u32x4;

union Frag { struct { u32x4 lo, hi; } u; v16bf v; };

__device__ __forceinline__ unsigned short f2bf(float f) {
  unsigned u = __float_as_uint(f);
  unsigned r = u + 0x7FFFu + ((u >> 16) & 1u);   // round-to-nearest-even
  return (unsigned short)(r >> 16);
}

// ---------------------------------------------------------------- init
__global__ void init_kernel(float* p) {
  if (threadIdx.x < 128) p[threadIdx.x] = 0.0f;
}

// ---------------------------------------------------------------- offset conv (18ch, 3x3, pad 1)
__global__ void __launch_bounds__(256) offset_conv_kernel(
    const float* __restrict__ x, const float* __restrict__ ow,
    const float* __restrict__ ob, float* __restrict__ off)
{
  __shared__ float sw[18 * Cc * Kk];                 // (oc, c, k) same order as input
  for (int i = threadIdx.x; i < 18 * Cc * Kk; i += 256) sw[i] = ow[i];
  __syncthreads();

  int idx  = blockIdx.x * 256 + threadIdx.x;
  int wcol = idx % Ww;
  int h    = (idx / Ww) % Hh;
  int b    = idx / HW;

  float acc[18];
  #pragma unroll
  for (int oc = 0; oc < 18; ++oc) acc[oc] = ob[oc];

  const float* xb = x + (size_t)b * Cc * HW;
  for (int kh = 0; kh < 3; ++kh) {
    int y = h - 1 + kh;
    if (y < 0 || y >= Hh) continue;
    for (int kw = 0; kw < 3; ++kw) {
      int xc = wcol - 1 + kw;
      if (xc < 0 || xc >= Ww) continue;
      int k = kh * 3 + kw;
      for (int c = 0; c < Cc; ++c) {
        float a = xb[c * HW + y * Ww + xc];
        #pragma unroll
        for (int oc = 0; oc < 18; ++oc)
          acc[oc] = fmaf(a, sw[(oc * Cc + c) * Kk + k], acc[oc]);
      }
    }
  }
  #pragma unroll
  for (int oc = 0; oc < 18; ++oc)
    off[((size_t)(b * 18 + oc)) * HW + h * Ww + wcol] = acc[oc];
}

// ---------------------------------------------------------------- deformable conv via bf16 WMMA
// 1 block = one (b,h) row (128 px) = 8 waves x 16-position tiles.
// GEMM: [16 pos x 288] x [288 x 64] with K chunked as (tap k)*32 + channel c.
__global__ void __launch_bounds__(256) deform_wmma_kernel(
    const float* __restrict__ x,  const float* __restrict__ off,
    const float* __restrict__ cw, const float* __restrict__ cb,
    float* __restrict__ out_pre, float* __restrict__ gsum, float* __restrict__ gsumsq)
{
  extern __shared__ unsigned short smem[];
  unsigned short* wl = smem;                  // weights: 64 rows x PITCH
  unsigned short* vl = smem + Oo * PITCH;     // vals   : 8*16 rows x PITCH

  const int tid  = threadIdx.x;
  const int wave = tid >> 5;
  const int lane = tid & 31;
  const int b    = blockIdx.x / Hh;
  const int h    = blockIdx.x % Hh;

  // Stage weights: wl[o][k*32+c] = cw[o][c][kh][kw]  (cw flat index == (o*32+c)*9+k)
  for (int i = tid; i < Oo * Cc * Kk; i += 256) {
    int k = i % Kk, c = (i / Kk) % Cc, o = i / (Kk * Cc);
    wl[o * PITCH + k * Cc + c] = f2bf(cw[i]);
  }

  // Gather bilinear samples for this wave's 16 positions.
  // lane = m + 16*chalf : position m, channel half chalf (16 ch each).
  const int m     = lane & 15;
  const int chalf = lane >> 4;
  const int wcol  = wave * 16 + m;
  const float* xb = x + (size_t)(b * Cc + chalf * 16) * HW;
  unsigned short* vrow = vl + (wave * 16 + m) * PITCH;

  for (int k = 0; k < Kk; ++k) {
    int kh = k / 3, kw = k % 3;
    float oy = off[((size_t)((b * Kk + k) * 2 + 0)) * HW + h * Ww + wcol];
    float ox = off[((size_t)((b * Kk + k) * 2 + 1)) * HW + h * Ww + wcol];
    float py = oy + (float)(h - 1 + kh);
    float px = ox + (float)(wcol - 1 + kw);
    float y0f = floorf(py), x0f = floorf(px);
    float wy = py - y0f, wx = px - x0f;
    int y0 = (int)y0f, x0i = (int)x0f;
    int y1 = y0 + 1, x1 = x0i + 1;
    // fold out-of-bounds validity into bilinear weights -> unconditional clamped loads
    float w00 = (1.f - wy) * (1.f - wx), w01 = (1.f - wy) * wx;
    float w10 = wy * (1.f - wx),         w11 = wy * wx;
    bool vy0 = (y0 >= 0) & (y0 < Hh), vy1 = (y1 >= 0) & (y1 < Hh);
    bool vx0 = (x0i >= 0) & (x0i < Ww), vx1 = (x1 >= 0) & (x1 < Ww);
    w00 = (vy0 && vx0) ? w00 : 0.f;  w01 = (vy0 && vx1) ? w01 : 0.f;
    w10 = (vy1 && vx0) ? w10 : 0.f;  w11 = (vy1 && vx1) ? w11 : 0.f;
    int cy0 = min(max(y0, 0), Hh - 1), cy1 = min(max(y1, 0), Hh - 1);
    int cx0 = min(max(x0i, 0), Ww - 1), cx1 = min(max(x1, 0), Ww - 1);
    int i00 = cy0 * Ww + cx0, i01 = cy0 * Ww + cx1;
    int i10 = cy1 * Ww + cx0, i11 = cy1 * Ww + cx1;
    #pragma unroll
    for (int ci = 0; ci < 16; ++ci) {
      const float* xc = xb + ci * HW;
      float v = w00 * xc[i00] + w01 * xc[i01] + w10 * xc[i10] + w11 * xc[i11];
      vrow[k * Cc + chalf * 16 + ci] = f2bf(v);
    }
  }
  __syncthreads();

  // WMMA: A = 16x32 bf16 (this wave's tile), B = 32x16 bf16 weight slices, C = f32.
  const int n    = lane & 15;
  const int half = lane >> 4;
  v8f z = {0.f, 0.f, 0.f, 0.f, 0.f, 0.f, 0.f, 0.f};
  v8f acc[4] = {z, z, z, z};
  const unsigned short* arow = vl + (wave * 16 + n) * PITCH;   // row M = lane%16

  #pragma unroll
  for (int k = 0; k < Kk; ++k) {
    // A layout (ISA 7.12.2, 16-bit 16x32): lanes 0-15 hold K{0-7,16-23}, lanes 16-31 K{8-15,24-31}
    Frag fa;
    const u32x4* pa = (const u32x4*)(arow + k * Cc);
    fa.u.lo = pa[half];
    fa.u.hi = pa[2 + half];
    #pragma unroll
    for (int t = 0; t < 4; ++t) {
      // B layout: lane -> column N = lane%16; lanes 0-15 K{0-7,16-23}, lanes 16-31 K{8-15,24-31} (matches A)
      Frag fb;
      const u32x4* pb = (const u32x4*)(wl + (t * 16 + n) * PITCH + k * Cc);
      fb.u.lo = pb[half];
      fb.u.hi = pb[2 + half];
      acc[t] = __builtin_amdgcn_wmma_f32_16x16x32_bf16(
          false, fa.v, false, fb.v, (short)0, acc[t], false, false);
    }
  }

  // Epilogue: bias, store pre-pool activations, per-channel sum / sumsq partials.
  // C/D layout: lane holds N = lane%16, rows M = j + 8*half for VGPR j.
  #pragma unroll
  for (int t = 0; t < 4; ++t) {
    int o = t * 16 + n;
    float bias = cb[o];
    float s = 0.f, s2 = 0.f;
    float* op = out_pre + ((size_t)(b * Oo + o) * Hh + h) * Ww + wave * 16 + half * 8;
    #pragma unroll
    for (int j = 0; j < 8; ++j) {
      float v = acc[t][j] + bias;
      op[j] = v;
      s += v;
      s2 += v * v;
    }
    atomicAdd(&gsum[o], s);
    atomicAdd(&gsumsq[o], s2);
  }
}

// ---------------------------------------------------------------- BN + ReLU + 2x2 maxpool
__global__ void __launch_bounds__(256) finalize_kernel(
    const float* __restrict__ out_pre, const float* __restrict__ gsum,
    const float* __restrict__ gsumsq, const float* __restrict__ gamma,
    const float* __restrict__ beta, float* __restrict__ out)
{
  int idx = blockIdx.x * 256 + threadIdx.x;
  int w2 = idx % (Ww / 2);
  int h2 = (idx / (Ww / 2)) % (Hh / 2);
  int o  = (idx / ((Ww / 2) * (Hh / 2))) % Oo;
  int b  = idx / ((Ww / 2) * (Hh / 2) * Oo);

  const float invN = 1.0f / (float)(Bn * HW);
  float mean = gsum[o] * invN;
  float var  = gsumsq[o] * invN - mean * mean;
  float inv  = rsqrtf(var + 1e-5f);
  float g  = gamma[o] * inv;
  float bb = beta[o] - mean * g;

  const float* p = out_pre + ((size_t)(b * Oo + o) * Hh + 2 * h2) * Ww + 2 * w2;
  float v0 = p[0] * g + bb, v1 = p[1] * g + bb;
  float v2 = p[Ww] * g + bb, v3 = p[Ww + 1] * g + bb;
  float m4 = fmaxf(fmaxf(v0, v1), fmaxf(v2, v3));
  out[idx] = fmaxf(m4, 0.0f);   // relu(max) == max(relu)
}

// ---------------------------------------------------------------- launch
extern "C" void kernel_launch(void* const* d_in, const int* in_sizes, int n_in,
                              void* d_out, int out_size, void* d_ws, size_t ws_size,
                              hipStream_t stream) {
  const float* x     = (const float*)d_in[0];
  const float* ow    = (const float*)d_in[1];  // offset_w
  const float* ob    = (const float*)d_in[2];  // offset_b
  // d_in[3]=mod_w, d_in[4]=mod_b: modulator is dead code in the reference output
  const float* cw    = (const float*)d_in[5];  // conv_w
  const float* cb    = (const float*)d_in[6];  // conv_b
  const float* gamma = (const float*)d_in[7];
  const float* beta  = (const float*)d_in[8];

  float* ws      = (float*)d_ws;
  float* gsum    = ws;                                  // 64
  float* gsumsq  = ws + 64;                             // 64
  float* off     = ws + 128;                            // B*18*H*W
  float* out_pre = off + (size_t)Bn * 18 * HW;          // B*64*H*W

  init_kernel<<<1, 128, 0, stream>>>(gsum);
  offset_conv_kernel<<<(Bn * HW) / 256, 256, 0, stream>>>(x, ow, ob, off);

  size_t lds_bytes = (size_t)(Oo + 8 * 16) * PITCH * sizeof(unsigned short);  // ~114 KB (<< 320 KB WGP LDS)
  deform_wmma_kernel<<<Bn * Hh, 256, lds_bytes, stream>>>(x, off, cw, cb, out_pre, gsum, gsumsq);

  finalize_kernel<<<(Bn * Oo * (Hh / 2) * (Ww / 2)) / 256, 256, 0, stream>>>(
      out_pre, gsum, gsumsq, gamma, beta, (float*)d_out);
}
